// SwichBackHead_57509612093558
// MI455X (gfx1250) — 8-bit matrix operand order repaired and checked by reading, compiled, not run on hardware
//
#include <hip/hip_runtime.h>
#include <stdint.h>

typedef int   v8i __attribute__((ext_vector_type(8)));
typedef int   v4i __attribute__((ext_vector_type(4)));
typedef int   v2i __attribute__((ext_vector_type(2)));
typedef float v4f __attribute__((ext_vector_type(4)));

#define HDIM 256
#define QMAX 127.0f

__global__ __launch_bounds__(256) void quant_rows_kernel(
    const float* __restrict__ src, int8_t* __restrict__ dstq,
    float* __restrict__ absmax_out, int nrows) {
  const int row  = (int)((blockIdx.x * blockDim.x + threadIdx.x) >> 5);
  const int lane = threadIdx.x & 31;
  if (row >= nrows) return;

  const v4f* rp = (const v4f*)(src + (size_t)row * HDIM);
  v4f a = __builtin_nontemporal_load(rp + lane * 2 + 0);
  v4f b = __builtin_nontemporal_load(rp + lane * 2 + 1);

  float m = fmaxf(fmaxf(fmaxf(fabsf(a.x), fabsf(a.y)), fmaxf(fabsf(a.z), fabsf(a.w))),
                  fmaxf(fmaxf(fabsf(b.x), fabsf(b.y)), fmaxf(fabsf(b.z), fabsf(b.w))));
  #pragma unroll
  for (int off = 16; off >= 1; off >>= 1)
    m = fmaxf(m, __shfl_xor(m, off, 32));

  const float scale = (m == 0.0f) ? 1.0f : m;
  const float k = QMAX / scale;

  int q0 = (int)rintf(a.x * k), q1 = (int)rintf(a.y * k);
  int q2 = (int)rintf(a.z * k), q3 = (int)rintf(a.w * k);
  int q4 = (int)rintf(b.x * k), q5 = (int)rintf(b.y * k);
  int q6 = (int)rintf(b.z * k), q7 = (int)rintf(b.w * k);

  v2i pk;
  pk.x = (q0 & 255) | ((q1 & 255) << 8) | ((q2 & 255) << 16) | (q3 << 24);
  pk.y = (q4 & 255) | ((q5 & 255) << 8) | ((q6 & 255) << 16) | (q7 << 24);
  *(v2i*)(dstq + (size_t)row * HDIM + lane * 8) = pk;

  if (lane == 0) absmax_out[row] = m;
}

__device__ __forceinline__ void gemm_tile_body(
    const int8_t* __restrict__ Xq, const int8_t* __restrict__ Wq,
    const float* __restrict__ sx, const float* __restrict__ sw,
    const float* __restrict__ bias, float* __restrict__ out,
    int N, int m0, int n0, int lane, int nsubs) {
  const int mrow = lane & 15;
  const int half = lane >> 4;

  const int8_t* arow = Xq + (size_t)(m0 + mrow) * HDIM + half * 16;
  v8i afrag[4];
  #pragma unroll
  for (int kc = 0; kc < 4; ++kc) {
    #pragma unroll
    for (int j = 0; j < 4; ++j) {
      v2i d = *(const v2i*)(arow + kc * 64 + j * 8 + (j >> 1) * 16);
      afrag[kc][2 * j + 0] = d.x;
      afrag[kc][2 * j + 1] = d.y;
    }
  }

  v8i acc[4];
  #pragma unroll
  for (int s = 0; s < 4; ++s) acc[s] = (v8i){0, 0, 0, 0, 0, 0, 0, 0};

  #pragma unroll
  for (int s = 0; s < 4; ++s) {
    if (s < nsubs) {
      const int8_t* brow = Wq + (size_t)(n0 + s * 16 + mrow) * HDIM + half * 16;
      #pragma unroll
      for (int kc = 0; kc < 4; ++kc) {
        v4i blo = *(const v4i*)(brow + kc * 64);
        v4i bhi = *(const v4i*)(brow + kc * 64 + 32);
        v8i bfrag;
        bfrag[0] = blo.x; bfrag[1] = blo.y; bfrag[2] = blo.z; bfrag[3] = blo.w;
        bfrag[4] = bhi.x; bfrag[5] = bhi.y; bfrag[6] = bhi.z; bfrag[7] = bhi.w;
        acc[s] = __builtin_amdgcn_wmma_i32_16x16x64_iu8(
             true, afrag[kc],  true, bfrag, acc[s],
             false,  false);
      }
    }
  }

  float sxv[8];
  #pragma unroll
  for (int r = 0; r < 8; ++r) sxv[r] = sx[m0 + r + half * 8];

  const float inv = 1.0f / (QMAX * QMAX);
  #pragma unroll
  for (int s = 0; s < 4; ++s) {
    if (s < nsubs) {
      const int n     = n0 + s * 16 + mrow;
      const float swn = sw[n] * inv;
      const float bn  = bias[n];
      #pragma unroll
      for (int r = 0; r < 8; ++r) {
        const int m = m0 + r + half * 8;
        __builtin_nontemporal_store((float)acc[s][r] * (sxv[r] * swn) + bn,
                                    &out[(size_t)m * N + n]);
      }
    }
  }
}

__global__ __launch_bounds__(256) void i8gemm_dequant_kernel(
    const int8_t* __restrict__ Xq, const int8_t* __restrict__ Wq,
    const float* __restrict__ sx, const float* __restrict__ sw,
    const float* __restrict__ bias, float* __restrict__ out,
    int T, int N) {
  const int lane = threadIdx.x & 31;
  const int wave = threadIdx.x >> 5;
  const int m0   = blockIdx.y * 16;
  const int n0   = blockIdx.x * 512 + wave * 64;
  if (n0 >= N) return;

  if (n0 + 64 <= N) {
    gemm_tile_body(Xq, Wq, sx, sw, bias, out, N, m0, n0, lane, 4);
  } else {
    gemm_tile_body(Xq, Wq, sx, sw, bias, out, N, m0, n0, lane, (N - n0) >> 4);
  }
}

extern "C" void kernel_launch(void* const* d_in, const int* in_sizes, int n_in,
                              void* d_out, int out_size, void* d_ws, size_t ws_size,
                              hipStream_t stream) {
  const float* X    = (const float*)d_in[0];
  const float* W    = (const float*)d_in[1];
  const float* bias = (const float*)d_in[2];
  float* out = (float*)d_out;

  const int T = in_sizes[0] / HDIM;
  const int N = in_sizes[2];

  uint8_t* ws = (uint8_t*)d_ws;
  size_t off = 0;
  int8_t* Xq = (int8_t*)(ws + off); off += (size_t)T * HDIM;  off = (off + 255) & ~(size_t)255;
  float*  sx = (float*)(ws + off);  off += (size_t)T * 4;     off = (off + 255) & ~(size_t)255;
  int8_t* Wq = (int8_t*)(ws + off); off += (size_t)N * HDIM;  off = (off + 255) & ~(size_t)255;
  float*  swv = (float*)(ws + off);

  quant_rows_kernel<<<(T + 7) / 8, 256, 0, stream>>>(X, Xq, sx, T);
  quant_rows_kernel<<<(N + 7) / 8, 256, 0, stream>>>(W, Wq, swv, N);

  dim3 grid((N + 511) / 512, T / 16);
  i8gemm_dequant_kernel<<<grid, 256, 0, stream>>>(Xq, Wq, sx, swv, bias, out, T, N);
}
